// WeightNet_72224170049803
// MI455X (gfx1250) — matrix operand order repaired and checked by reading, not compiled, not run on hardware
//
#include <hip/hip_runtime.h>
#include <hip/hip_bf16.h>
#include <math.h>

typedef __attribute__((ext_vector_type(16))) __bf16 bf16x16;
typedef __attribute__((ext_vector_type(8)))  float  f32x8;
typedef __attribute__((ext_vector_type(4)))  int    i32x4;

#define INP_C 256
#define OUP_C 256
#define NB    32
#define HDIM  32
#define WDIM  32
#define HW    1024
#define MOUP  512

#ifndef __has_builtin
#define __has_builtin(x) 0
#endif
#if __has_builtin(__builtin_amdgcn_global_load_async_to_lds_b128)
#define WN_ASYNC 1
#else
#define WN_ASYNC 0
#endif

__device__ __forceinline__ unsigned short f2bf(float f) {
    unsigned int u = __float_as_uint(f);
    u += 0x7FFFu + ((u >> 16) & 1u);          // round-to-nearest-even
    return (unsigned short)(u >> 16);
}

#if WN_ASYNC
// 32-byte global->LDS async copy: two b128, ISA offset applies to both sides.
__device__ __forceinline__ void async_cp32(const void* g, void* l) {
    auto gp = (__attribute__((address_space(1))) i32x4*)(uintptr_t)g;
    auto lp = (__attribute__((address_space(3))) i32x4*)(unsigned int)(uintptr_t)l;
    __builtin_amdgcn_global_load_async_to_lds_b128(gp, lp, 0, 0);
    __builtin_amdgcn_global_load_async_to_lds_b128(gp, lp, 16, 0);
}
__device__ __forceinline__ void wait_async0() {
#if __has_builtin(__builtin_amdgcn_s_wait_asynccnt)
    __builtin_amdgcn_s_wait_asynccnt(0);
#else
    asm volatile("s_wait_asynccnt 0x0" ::: "memory");
#endif
}
#endif

// ---------------- 1) global average pool: gap[b,c] = mean over 32x32 ----------
__global__ __launch_bounds__(256) void wn_gap(const float* __restrict__ x,
                                              float* __restrict__ gap) {
    __shared__ float red[256];
    const int t = threadIdx.x;
    const float* p = x + (size_t)blockIdx.x * HW;
    float s = p[t] + p[t + 256] + p[t + 512] + p[t + 768];
    red[t] = s;
    __syncthreads();
    for (int off = 128; off > 32; off >>= 1) {
        if (t < off) red[t] += red[t + off];
        __syncthreads();
    }
    if (t < 32) {
        float v = red[t] + red[t + 32];
        for (int off = 16; off > 0; off >>= 1) v += __shfl_down(v, off, 32);
        if (t == 0) gap[blockIdx.x] = v * (1.0f / 1024.0f);
    }
}

// ---------------- 2) tiny MLP: g = gap@Wr^T+br ; a = sigmoid(g@W1^T+b1) -------
__global__ __launch_bounds__(512) void wn_mlp(const float* __restrict__ gap,
                                              const float* __restrict__ Wr,
                                              const float* __restrict__ br,
                                              const float* __restrict__ W1,
                                              const float* __restrict__ b1,
                                              float* __restrict__ a) {
    __shared__ float gsh[NB * 16];
    const int t = threadIdx.x;                 // 0..511
    {
        const int bb = t >> 4, j = t & 15;
        float s = br[j];
        const float* gp = gap + bb * INP_C;
        const float* wp = Wr + j * INP_C;
        for (int i = 0; i < INP_C; ++i) s += gp[i] * wp[i];
        gsh[t] = s;
    }
    __syncthreads();
    {
        const float bias = b1[t];
        for (int bb = 0; bb < NB; ++bb) {
            float s = bias;
            const float* gp = gsh + bb * 16;
            const float* wp = W1 + t * 16;
            #pragma unroll
            for (int j = 0; j < 16; ++j) s += gp[j] * wp[j];
            a[bb * MOUP + t] = 1.0f / (1.0f + __expf(-s));
        }
    }
}

// ---------------- 3) materialize per-sample bf16 weights: Wb[b][r][o][i] ------
__global__ __launch_bounds__(256) void wn_wprep(const float* __restrict__ a,
                                                const float* __restrict__ Wfc2,
                                                unsigned short* __restrict__ Wb) {
    const unsigned int idx = blockIdx.x * 256u + threadIdx.x;   // < 32*9*65536
    const unsigned int i  = idx & 255u;
    const unsigned int o  = (idx >> 8) & 255u;
    const unsigned int br = idx >> 16;
    const unsigned int r  = br % 9u;
    const unsigned int b  = br / 9u;
    const float av = a[b * MOUP + 2u * o + (i >= 128u ? 1u : 0u)];
    const float wv = Wfc2[((o << 8) + i) * 9u + r];
    Wb[idx] = f2bf(av * wv);
}

// ---------------- 3b) transpose + convert x: xT[b][hw][i] = bf16(x[b][i][hw]) -
__global__ __launch_bounds__(256) void wn_xprep(const float* __restrict__ x,
                                                unsigned short* __restrict__ xT) {
    __shared__ unsigned short tile[64][65];
    const int n0 = blockIdx.x << 6;
    const int i0 = blockIdx.y << 6;
    const int b  = blockIdx.z;
    const float* xb = x + ((size_t)b * INP_C + i0) * HW + n0;
    unsigned short* xTb = xT + ((size_t)b * HW + n0) * INP_C + i0;
    const int t = threadIdx.x;
    #pragma unroll
    for (int rep = 0; rep < 16; ++rep) {
        const int idx = rep * 256 + t;
        const int ii = idx >> 6, nn = idx & 63;
        tile[ii][nn] = f2bf(xb[(size_t)ii * HW + nn]);   // coalesced along n
    }
    __syncthreads();
    #pragma unroll
    for (int rep = 0; rep < 16; ++rep) {
        const int idx = rep * 256 + t;
        const int nn = idx >> 6, ii = idx & 63;
        xTb[(size_t)nn * INP_C + ii] = tile[ii][nn];     // coalesced along i
    }
}

// ---------------- 4) implicit-GEMM conv via bf16 WMMA -------------------------
// Per sample: C[256 o, 1024 hw] = sum over 9 taps x 256 ch of A[o,i]*B[i,hw].
// 128(o) x 128(hw) macro-tile, double-buffered 32-K stages staged with async
// global->LDS copies, 8 wave32s, each wave owns 2x4 sixteen-square C tiles.
__global__ __launch_bounds__(256) void wn_conv_wmma(
        const unsigned short* __restrict__ xT,       // [32,1024,256] bf16
        const unsigned short* __restrict__ Wb,       // [32,9,256,256] bf16
        float* __restrict__ y) {                     // [32,256,32,32] f32
    __shared__ unsigned short Alds[2][128][32];      // [buf][o_local][i_local]
    __shared__ unsigned short Blds[2][128][32];      // [buf][n_local][i_local]

    const int n0 = blockIdx.x * 128;                 // spatial tile (4 rows)
    const int o0 = blockIdx.y * 128;                 // out-channel tile
    const int b  = blockIdx.z;

    const int t    = threadIdx.x;
    const int wave = t >> 5;
    const int lane = t & 31;
    const int l16  = lane & 15;
    const bool hi  = lane >= 16;

    const int mi0 = (wave & 3) << 1;                 // 2 m-subtiles per wave
    const int nh  = (wave >> 2) << 6;                // 0 or 64 col offset

    f32x8 acc[2][4];
    #pragma unroll
    for (int m = 0; m < 2; ++m)
        #pragma unroll
        for (int j = 0; j < 4; ++j)
            acc[m][j] = (f32x8){0.f, 0.f, 0.f, 0.f, 0.f, 0.f, 0.f, 0.f};

    // staging assignment: 256 threads -> 128 rows x 2 sixteen-half chunks
    const int s_row = t >> 1;                        // 0..127
    const int s_kb  = (t & 1) << 4;                  // 0 or 16 halves
    const int hb = (n0 + s_row) >> 5;
    const int wb = (n0 + s_row) & 31;

    const unsigned short* xTb = xT + (size_t)b * HW * INP_C;
    const unsigned short* WbB = Wb + (size_t)b * 9 * OUP_C * INP_C;

    const int aklo = hi ? 8 : 0;
    const int bk0  = hi ? 8 : 0;

    // Issue one 32-K stage (tap r = s/8, channels (s%8)*32..+31) into buffer nbuf.
    auto stage_issue = [&](int s, int nbuf) {
        const int r  = s >> 3;
        const int i0 = (s & 7) << 5;
        const unsigned short* Wsrc =
            WbB + ((size_t)r * OUP_C + o0 + s_row) * INP_C + i0 + s_kb;
        unsigned short* adst = &Alds[nbuf][s_row][s_kb];
        const int kh = r / 3 - 1, kw = r % 3 - 1;
        const int hs = hb + kh, wsr = wb + kw;
        const bool inb = (hs >= 0) & (hs < HDIM) & (wsr >= 0) & (wsr < WDIM);
        const unsigned short* Bsrc =
            xTb + ((size_t)(hs * WDIM + wsr)) * INP_C + i0 + s_kb;
        unsigned short* bdst = &Blds[nbuf][s_row][s_kb];
#if WN_ASYNC
        async_cp32(Wsrc, adst);
        if (inb) {
            async_cp32(Bsrc, bdst);
        } else {
            const uint4 z = make_uint4(0u, 0u, 0u, 0u);
            *(uint4*)bdst = z;
            *(uint4*)(bdst + 8) = z;
        }
#else
        *(uint4*)adst       = *(const uint4*)Wsrc;
        *(uint4*)(adst + 8) = *(const uint4*)(Wsrc + 8);
        uint4 v0 = make_uint4(0u, 0u, 0u, 0u), v1 = v0;
        if (inb) { v0 = *(const uint4*)Bsrc; v1 = *(const uint4*)(Bsrc + 8); }
        *(uint4*)bdst       = v0;
        *(uint4*)(bdst + 8) = v1;
#endif
    };

    stage_issue(0, 0);
#if WN_ASYNC
    wait_async0();
#endif
    __syncthreads();

    for (int s = 0; s < 72; ++s) {                   // 9 taps x 8 k-chunks
        const int buf = s & 1;
        if (s < 71) stage_issue(s + 1, buf ^ 1);     // async fill of back buffer

        union F { bf16x16 v; uint4 q[2]; };
        F fb[4];
        #pragma unroll
        for (int j = 0; j < 4; ++j) {
            const int brow = nh + (j << 4) + l16;
            fb[j].q[0] = *(const uint4*)&Blds[buf][brow][bk0];
            fb[j].q[1] = *(const uint4*)&Blds[buf][brow][bk0 + 16];
        }
        F fa[2];
        #pragma unroll
        for (int m = 0; m < 2; ++m) {
            const int arow = ((mi0 + m) << 4) + l16;
            fa[m].q[0] = *(const uint4*)&Alds[buf][arow][aklo];
            fa[m].q[1] = *(const uint4*)&Alds[buf][arow][aklo + 16];
        }
        #pragma unroll
        for (int m = 0; m < 2; ++m)
            #pragma unroll
            for (int j = 0; j < 4; ++j)
                acc[m][j] = __builtin_amdgcn_wmma_f32_16x16x32_bf16(
                    false, fa[m].v, false, fb[j].v, (short)0, acc[m][j],
                    false, false);

#if WN_ASYNC
        if (s < 71) wait_async0();                   // copies hidden under WMMAs
#endif
        __syncthreads();
    }

    // C layout: VGPR g, lanes 0-15 -> M=g, lanes 16-31 -> M=g+8, N=l16
    float* yb = y + (size_t)b * OUP_C * HW;
    #pragma unroll
    for (int m = 0; m < 2; ++m) {
        const int orow = o0 + ((mi0 + m) << 4) + (hi ? 8 : 0);
        #pragma unroll
        for (int j = 0; j < 4; ++j) {
            const int ncol = n0 + nh + (j << 4) + l16;
            #pragma unroll
            for (int g = 0; g < 8; ++g)
                yb[(size_t)(orow + g) * HW + ncol] = acc[m][j][g];
        }
    }
}

extern "C" void kernel_launch(void* const* d_in, const int* in_sizes, int n_in,
                              void* d_out, int out_size, void* d_ws, size_t ws_size,
                              hipStream_t stream) {
    const float* x        = (const float*)d_in[0];
    const float* W_reduce = (const float*)d_in[1];
    const float* b_reduce = (const float*)d_in[2];
    const float* W_fc1    = (const float*)d_in[3];
    const float* b_fc1    = (const float*)d_in[4];
    const float* W_fc2    = (const float*)d_in[5];
    float* y = (float*)d_out;

    char* ws = (char*)d_ws;
    float* gap = (float*)ws;                                   //  32 KB
    float* a   = (float*)(ws + (32 << 10));                    //  64 KB
    unsigned short* Wb = (unsigned short*)(ws + (128 << 10));  // 36 MB bf16 weights
    unsigned short* xT = (unsigned short*)(ws + (128 << 10) + (size_t)NB * 9 * OUP_C * INP_C * 2);

    wn_gap<<<NB * INP_C, 256, 0, stream>>>(x, gap);
    wn_mlp<<<1, 512, 0, stream>>>(gap, W_reduce, b_reduce, W_fc1, b_fc1, a);
    wn_wprep<<<(NB * 9 * OUP_C * INP_C) / 256, 256, 0, stream>>>(a, W_fc2, Wb);
    {
        dim3 g(HW / 64, INP_C / 64, NB);   // 16 x 4 x 32
        wn_xprep<<<g, 256, 0, stream>>>(x, xT);
    }
    dim3 grid(HW / 128, OUP_C / 128, NB);  // 8 x 2 x 32
    wn_conv_wmma<<<grid, 256, 0, stream>>>(xT, Wb, y);
}
